// MLA_28793460752680
// MI455X (gfx1250) — compile-verified, matrix operand order checked by reading, not run on hardware
//
#include <hip/hip_runtime.h>

// ---------------------------------------------------------------------------
// MLA forward for MI455X (gfx1250, wave32, WMMA f32<-f16), f16 data paths
// B=2, T=2048, C=1024, NH=16, LAT=512, DHR=64, DK=64, M=B*T=4096
// ---------------------------------------------------------------------------

typedef __attribute__((ext_vector_type(16))) _Float16 v16h;
typedef __attribute__((ext_vector_type(8)))  _Float16 h8;
typedef __attribute__((ext_vector_type(2)))  _Float16 h2;
typedef __attribute__((ext_vector_type(8)))  float    v8f;

#define MLA_B   2
#define MLA_T   2048
#define MLA_C   1024
#define MLA_NH  16
#define MLA_M   (MLA_B * MLA_T)   // 4096

// ---- WMMA wrapper ---------------------------------------------------------
__device__ __forceinline__ v8f wmma_f16(v16h a, v16h b, v8f c) {
  // (neg_a, A, neg_b, B, c_mod, C, reuse_a, reuse_b)
  return __builtin_amdgcn_wmma_f32_16x16x32_f16(false, a, false, b,
                                                (short)0, c, false, false);
}

// ---- Fragment loader (CDNA5 16-bit A/B layout, wave32) --------------------
// lane L<16 holds row L, K = {kb..kb+7, kb+16..kb+23} with kb = (lane>>4)*8.
// Vector elements 2v,2v+1 land in VGPR v halves. Two b128 loads, no converts.
__device__ __forceinline__ v16h frag_ld_f16(const _Float16* __restrict__ base,
                                            int ld) {
  int lane = threadIdx.x & 31;
  const _Float16* p = base + (size_t)(lane & 15) * ld + ((lane >> 4) << 3);
  h8 lo = *(const h8*)(p);
  h8 hi = *(const h8*)(p + 16);
  v16h f;
#pragma unroll
  for (int i = 0; i < 8; ++i) { f[i] = lo[i]; f[8 + i] = hi[i]; }
  return f;
}

// ---------------------------------------------------------------------------
// GEMM: C[M,N] = A_f16[M,K](lda) @ Wt_f16[N,K] + bias[N]
// Block = 128 threads (4 waves, 2x2), block tile 128x64, wave tile 64x32:
// 6 fragment loads -> 8 WMMAs per k-step. fp32 accumulate, f16 or f32 store.
// Weights sit in the 192MB L2; A streams from HBM as f16.
// ---------------------------------------------------------------------------
__global__ __launch_bounds__(128) void gemm_f16a_f16w(
    const _Float16* __restrict__ A, int lda,
    const _Float16* __restrict__ Wt,
    const float* __restrict__ bias,
    _Float16* __restrict__ Ch, float* __restrict__ Cf,
    int M, int N, int K) {
  int wave = threadIdx.x >> 5;
  int lane = threadIdx.x & 31;
  int nn = lane & 15, hh = lane >> 4;
  int row0 = blockIdx.y * 128 + (wave >> 1) * 64;
  int col0 = blockIdx.x * 64 + (wave & 1) * 32;

  v8f acc[4][2] = {};
#pragma unroll 2
  for (int k = 0; k < K; k += 32) {
    if (k + 128 < K) {
      __builtin_prefetch(A + (size_t)row0 * lda + k + 128);
      __builtin_prefetch(A + (size_t)(row0 + 32) * lda + k + 128);
    }
    v16h a[4];
#pragma unroll
    for (int r = 0; r < 4; ++r)
      a[r] = frag_ld_f16(A + (size_t)(row0 + 16 * r) * lda + k, lda);
    v16h b0 = frag_ld_f16(Wt + (size_t)col0 * K + k, K);
    v16h b1 = frag_ld_f16(Wt + (size_t)(col0 + 16) * K + k, K);
#pragma unroll
    for (int r = 0; r < 4; ++r) {
      acc[r][0] = wmma_f16(a[r], b0, acc[r][0]);
      acc[r][1] = wmma_f16(a[r], b1, acc[r][1]);
    }
  }
#pragma unroll
  for (int mi = 0; mi < 4; ++mi)
#pragma unroll
    for (int ni = 0; ni < 2; ++ni) {
      int n = col0 + ni * 16 + nn;
      float bv = bias[n];
#pragma unroll
      for (int i = 0; i < 8; ++i) {
        int m = row0 + mi * 16 + i + 8 * hh;
        float v = acc[mi][ni][i] + bv;
        if (Cf) Cf[(size_t)m * N + n] = v;
        else    Ch[(size_t)m * N + n] = (_Float16)v;
      }
    }
}

// ---- f32 -> f16 convert (for x) -------------------------------------------
__global__ void f32_to_f16(const float* __restrict__ s,
                           _Float16* __restrict__ d, int n) {
  int i = blockIdx.x * blockDim.x + threadIdx.x;
  if (i < n) d[i] = (_Float16)s[i];
}

// ---------------------------------------------------------------------------
// In-place RoPE on f16 buffer, f32 math; matches reference _rope
// (freq = t+1, theta = 1e4^(-2p/d))
// ---------------------------------------------------------------------------
__global__ void rope_f16(_Float16* __restrict__ x, int rows, int d) {
  int half = d >> 1;
  int idx = blockIdx.x * blockDim.x + threadIdx.x;
  if (idx >= rows * half) return;
  int r = idx / half, p = idx - r * half;
  int t = r & (MLA_T - 1);
  float theta = __expf((-2.0f * (float)p / (float)d) * 9.2103403719761836f);
  float ang = (float)(t + 1) * theta;
  float sn, cs;
  __sincosf(ang, &sn, &cs);
  h2* q = (h2*)(x + (size_t)r * d + 2 * p);
  h2 v = *q;
  float x1 = (float)v[0], x2 = (float)v[1];
  v[0] = (_Float16)(x1 * cs - x2 * sn);
  v[1] = (_Float16)(x2 * cs + x1 * sn);
  *q = v;
}

// ---- weight transpose + f32->f16 ------------------------------------------
__global__ void transpose_to_f16(const float* __restrict__ src,
                                 _Float16* __restrict__ dst, int K, int N) {
  int idx = blockIdx.x * blockDim.x + threadIdx.x;
  if (idx >= K * N) return;
  int k = idx / N, n = idx - k * N;
  dst[(size_t)n * K + k] = (_Float16)src[idx];
}

// ---- assemble Q/K heads: [B,NH,T,128] = concat(nope, rope), Q pre-scaled --
__global__ void assemble_qk(const _Float16* __restrict__ qlin,
                            const _Float16* __restrict__ qr,
                            const _Float16* __restrict__ kv,
                            const _Float16* __restrict__ kr,
                            _Float16* __restrict__ Qf,
                            _Float16* __restrict__ Kf) {
  int idx = blockIdx.x * blockDim.x + threadIdx.x;  // B*NH*T*128 = 2^23
  if (idx >= MLA_B * MLA_NH * MLA_T * 128) return;
  int d = idx & 127;
  int t = (idx >> 7) & (MLA_T - 1);
  int h = (idx >> 18) & (MLA_NH - 1);
  int b = idx >> 22;
  size_t row = (size_t)(b * MLA_T + t);
  float qv = (float)((d < 64) ? qlin[row * 1024 + h * 64 + d]
                              : qr[row * 1024 + h * 64 + (d - 64)]);
  _Float16 kvv = (d < 64) ? kv[row * 2048 + h * 64 + d]
                          : kr[row * 64 + (d - 64)];
  size_t o = ((size_t)(b * MLA_NH + h) * MLA_T + t) * 128 + d;
  Qf[o] = (_Float16)(qv * 0.08838834764831843f);  // 1/sqrt(128)
  Kf[o] = kvv;
}

// ---- assemble V^T: [B,NH,64,T] --------------------------------------------
__global__ void assemble_vt(const _Float16* __restrict__ kv,
                            _Float16* __restrict__ Vt) {
  int idx = blockIdx.x * blockDim.x + threadIdx.x;  // B*NH*T*64 = 2^22
  if (idx >= MLA_B * MLA_NH * MLA_T * 64) return;
  int d = idx & 63;
  int t = (idx >> 6) & (MLA_T - 1);
  int h = (idx >> 17) & (MLA_NH - 1);
  int b = idx >> 21;
  Vt[((size_t)(b * MLA_NH + h) * 64 + d) * MLA_T + t] =
      kv[(size_t)(b * MLA_T + t) * 2048 + 1024 + h * 64 + d];
}

// ---------------------------------------------------------------------------
// Flash attention: causal, head dim 128 (QK), 64 (V). Block = 4 waves,
// each wave owns a 16-row query tile, streams 32-key tiles.
// Online softmax via xor-shuffles over the 16-lane halves (C-layout rows).
// P C-layout -> A-fragment via per-wave LDS tile (in-order DS, no barrier).
// Output written as f16 (consumed by the f16-A final GEMM).
// ---------------------------------------------------------------------------
__global__ __launch_bounds__(128) void mla_flash_attn(
    const _Float16* __restrict__ Q, const _Float16* __restrict__ Kf,
    const _Float16* __restrict__ Vt, _Float16* __restrict__ out) {
  __shared__ __align__(64) _Float16 ptile[4][16 * 32];
  int wave = threadIdx.x >> 5;
  int lane = threadIdx.x & 31;
  int nn = lane & 15, hh = lane >> 4;
  int b = blockIdx.z, h = blockIdx.y;
  int qbase = blockIdx.x * 64 + wave * 16;

  const _Float16* Qbh = Q  + (size_t)(b * MLA_NH + h) * MLA_T * 128;
  const _Float16* Kbh = Kf + (size_t)(b * MLA_NH + h) * MLA_T * 128;
  const _Float16* Vbh = Vt + (size_t)(b * MLA_NH + h) * 64 * MLA_T;

  v16h aq[4];
#pragma unroll
  for (int c = 0; c < 4; ++c)
    aq[c] = frag_ld_f16(Qbh + (size_t)qbase * 128 + c * 32, 128);

  v8f o[4] = {};
  float mrun[8], lrun[8];
#pragma unroll
  for (int i = 0; i < 8; ++i) { mrun[i] = -1e30f; lrun[i] = 0.0f; }

  int ntiles = (qbase + 16 + 31) >> 5;
  for (int j = 0; j < ntiles; ++j) {
    int kt0 = j * 32;
    v8f s0 = {}, s1 = {};
#pragma unroll
    for (int c = 0; c < 4; ++c) {
      v16h bk0 = frag_ld_f16(Kbh + (size_t)kt0 * 128 + c * 32, 128);
      v16h bk1 = frag_ld_f16(Kbh + (size_t)(kt0 + 16) * 128 + c * 32, 128);
      s0 = wmma_f16(aq[c], bk0, s0);
      s1 = wmma_f16(aq[c], bk1, s1);
    }
    if (kt0 + 31 > qbase) {  // diagonal tile: causal mask
#pragma unroll
      for (int i = 0; i < 8; ++i) {
        int qg = qbase + i + 8 * hh;
        s0[i] = (kt0 + nn > qg) ? -1e30f : s0[i];
        s1[i] = (kt0 + 16 + nn > qg) ? -1e30f : s1[i];
      }
    }
    _Float16* pw = ptile[wave];
#pragma unroll
    for (int i = 0; i < 8; ++i) {
      float v = fmaxf(s0[i], s1[i]);
      v = fmaxf(v, __shfl_xor(v, 1, 32));
      v = fmaxf(v, __shfl_xor(v, 2, 32));
      v = fmaxf(v, __shfl_xor(v, 4, 32));
      v = fmaxf(v, __shfl_xor(v, 8, 32));
      float mn = fmaxf(mrun[i], v);
      float sc = __expf(mrun[i] - mn);
      float p0 = __expf(s0[i] - mn);
      float p1 = __expf(s1[i] - mn);
      float rs = p0 + p1;
      rs += __shfl_xor(rs, 1, 32);
      rs += __shfl_xor(rs, 2, 32);
      rs += __shfl_xor(rs, 4, 32);
      rs += __shfl_xor(rs, 8, 32);
      lrun[i] = lrun[i] * sc + rs;
      mrun[i] = mn;
#pragma unroll
      for (int c = 0; c < 4; ++c) o[c][i] *= sc;
      int r = i + 8 * hh;
      pw[r * 32 + nn]      = (_Float16)p0;
      pw[r * 32 + 16 + nn] = (_Float16)p1;
    }
    asm volatile("s_wait_dscnt 0" ::: "memory");  // DS in-order per wave; insurance
    v16h ap = frag_ld_f16(pw, 32);
#pragma unroll
    for (int c = 0; c < 4; ++c) {
      v16h bv = frag_ld_f16(Vbh + (size_t)(c * 16) * MLA_T + kt0, MLA_T);
      o[c] = wmma_f16(ap, bv, o[c]);
    }
  }

  float inv[8];
#pragma unroll
  for (int i = 0; i < 8; ++i) inv[i] = 1.0f / lrun[i];
#pragma unroll
  for (int c = 0; c < 4; ++c)
#pragma unroll
    for (int i = 0; i < 8; ++i) {
      size_t row = (size_t)(b * MLA_T + qbase + i + 8 * hh);
      out[row * MLA_C + h * 64 + c * 16 + nn] = (_Float16)(o[c][i] * inv[i]);
    }
}

// ---------------------------------------------------------------------------
extern "C" void kernel_launch(void* const* d_in, const int* in_sizes, int n_in,
                              void* d_out, int out_size, void* d_ws,
                              size_t ws_size, hipStream_t stream) {
  (void)in_sizes; (void)n_in; (void)out_size; (void)ws_size;
  const float* x   = (const float*)d_in[0];
  const float* w1  = (const float*)d_in[1];
  const float* b1  = (const float*)d_in[2];
  const float* wkr = (const float*)d_in[3];
  const float* bkr = (const float*)d_in[4];
  const float* wqr = (const float*)d_in[5];
  const float* bqr = (const float*)d_in[6];
  const float* wkv = (const float*)d_in[7];
  const float* bkv = (const float*)d_in[8];
  const float* wq  = (const float*)d_in[9];
  const float* bq  = (const float*)d_in[10];
  const float* wo  = (const float*)d_in[11];
  const float* bo  = (const float*)d_in[12];
  float* out = (float*)d_out;

  char* base = (char*)d_ws;
  size_t off = 0;
  auto alloc = [&](size_t elems, size_t esz) -> void* {
    void* p = base + off;
    off += (elems * esz + 255) & ~(size_t)255;
    return p;
  };

  _Float16* w1t   = (_Float16*)alloc(1024 * 1024, 2);
  _Float16* wkrt  = (_Float16*)alloc(1024 * 64, 2);
  _Float16* wqrt  = (_Float16*)alloc(1024 * 1024, 2);
  _Float16* wkvt  = (_Float16*)alloc(512 * 2048, 2);
  _Float16* wqt   = (_Float16*)alloc(512 * 1024, 2);
  _Float16* wot   = (_Float16*)alloc(1024 * 1024, 2);
  _Float16* xh    = (_Float16*)alloc((size_t)MLA_M * 1024, 2);
  _Float16* hbuf  = (_Float16*)alloc((size_t)MLA_M * 1024, 2);
  _Float16* krlin = (_Float16*)alloc((size_t)MLA_M * 64, 2);
  _Float16* qrlin = (_Float16*)alloc((size_t)MLA_M * 1024, 2);
  _Float16* kvb   = (_Float16*)alloc((size_t)MLA_M * 2048, 2);
  _Float16* qlin  = (_Float16*)alloc((size_t)MLA_M * 1024, 2);
  _Float16* Qf    = (_Float16*)alloc((size_t)MLA_B * MLA_NH * MLA_T * 128, 2);
  _Float16* Kff   = (_Float16*)alloc((size_t)MLA_B * MLA_NH * MLA_T * 128, 2);
  _Float16* Vtb   = (_Float16*)alloc((size_t)MLA_B * MLA_NH * 64 * MLA_T, 2);
  _Float16* attn  = (_Float16*)alloc((size_t)MLA_M * 1024, 2);

  auto tl = [&](const float* s, _Float16* d, int K, int N) {
    int tot = K * N;
    transpose_to_f16<<<(tot + 255) / 256, 256, 0, stream>>>(s, d, K, N);
  };
  tl(w1, w1t, 1024, 1024);
  tl(wkr, wkrt, 1024, 64);
  tl(wqr, wqrt, 1024, 1024);
  tl(wkv, wkvt, 512, 2048);
  tl(wq, wqt, 512, 1024);
  tl(wo, wot, 1024, 1024);

  f32_to_f16<<<(MLA_M * 1024 + 255) / 256, 256, 0, stream>>>(
      x, xh, MLA_M * 1024);

  auto gemm = [&](const _Float16* A, int lda, const _Float16* Wt,
                  const float* bias, _Float16* Ch, float* Cf, int N, int K) {
    dim3 g(N / 64, MLA_M / 128);
    gemm_f16a_f16w<<<g, 128, 0, stream>>>(A, lda, Wt, bias, Ch, Cf,
                                          MLA_M, N, K);
  };

  // h = x @ w1 + b1
  gemm(xh, 1024, w1t, b1, hbuf, nullptr, 1024, 1024);
  // kR/qR pre-rope projections (from full h)
  gemm(hbuf, 1024, wkrt, bkr, krlin, nullptr, 64, 1024);
  gemm(hbuf, 1024, wqrt, bqr, qrlin, nullptr, 1024, 1024);
  // kv = cKV @ wkv + bkv   (cKV = h[:, :512])
  gemm(hbuf, 1024, wkvt, bkv, kvb, nullptr, 2048, 512);
  // q = cq @ wq + bq       (cq = h[:, 512:])
  gemm(hbuf + 512, 1024, wqt, bq, qlin, nullptr, 1024, 512);

  rope_f16<<<((MLA_M * 32) + 255) / 256, 256, 0, stream>>>(krlin, MLA_M, 64);
  rope_f16<<<((MLA_M * 512) + 255) / 256, 256, 0, stream>>>(qrlin, MLA_M, 1024);

  assemble_qk<<<(MLA_B * MLA_NH * MLA_T * 128) / 256, 256, 0, stream>>>(
      qlin, qrlin, kvb, krlin, Qf, Kff);
  assemble_vt<<<(MLA_B * MLA_NH * MLA_T * 64) / 256, 256, 0, stream>>>(kvb, Vtb);

  mla_flash_attn<<<dim3(MLA_T / 64, MLA_NH, MLA_B), 128, 0, stream>>>(
      Qf, Kff, Vtb, attn);

  // out = attn @ wo + bo
  gemm(attn, 1024, wot, bo, nullptr, out, 1024, 1024);
}
